// KermtAutoregressive_68702296867113
// MI455X (gfx1250) — compile-verified, matrix operand order checked by reading, not run on hardware
//
#include <hip/hip_runtime.h>
#include <hip/hip_bf16.h>
#include <math.h>

#define TT 2048
#define MD 1024            // M == H == 1024
#define NSOL 5
#define DE 64
#define NB 3
#define MM ((size_t)MD * MD)
#define NWG 64             // persistent workgroups in the scan
#define OPW 16             // output rows owned per workgroup (MD / NWG)

typedef __bf16 bf16;
typedef __attribute__((ext_vector_type(16))) __bf16 v16bf;
typedef __attribute__((ext_vector_type(8)))  __bf16 v8bf;
typedef __attribute__((ext_vector_type(8)))  float  v8f;

__device__ __forceinline__ float geluf(float x) {
    return 0.5f * x * (1.0f + erff(x * 0.7071067811865476f));
}
// device-scope (agent) loads/stores: bypass stale WGP$ lines for cross-WG data
__device__ __forceinline__ float gload(const float* p) {
    return __hip_atomic_load(p, __ATOMIC_RELAXED, __HIP_MEMORY_SCOPE_AGENT);
}
__device__ __forceinline__ void gstore(float* p, float v) {
    __hip_atomic_store(p, v, __ATOMIC_RELAXED, __HIP_MEMORY_SCOPE_AGENT);
}

// ---------------------------------------------------------------------------
// fp32 [K=1024,N=1024] -> bf16 [N,K] transposed (for WMMA B operands), LDS tiled
// ---------------------------------------------------------------------------
__global__ void k_cvt_t(const float* __restrict__ src, bf16* __restrict__ dst) {
    __shared__ float tile[32][33];
    int bx = blockIdx.x * 32, by = blockIdx.y * 32;
    int tx = threadIdx.x, ty = threadIdx.y;       // block = (32, 8)
    for (int r = ty; r < 32; r += 8)
        tile[r][tx] = src[(size_t)(by + r) * MD + (bx + tx)];
    __syncthreads();
    for (int r = ty; r < 32; r += 8)
        dst[(size_t)(bx + r) * MD + (by + tx)] = (bf16)tile[tx][r];
}

// ---------------------------------------------------------------------------
// Scan-permuted weight layout: dst[o*1024 + sub*64 + j] = src[(sub+16j)*MD + o]
// Thread (o,sub) in the scan then loads 64 contiguous bf16 while its LDS
// activation reads (k = sub+16j) hit 16 distinct banks.
// ---------------------------------------------------------------------------
__global__ void k_cvt_perm(const float* __restrict__ src, bf16* __restrict__ dst) {
    int idx = blockIdx.x * blockDim.x + threadIdx.x;
    int o = idx >> 10, r = idx & 1023;
    int sub = r >> 6, j = r & 63;
    int k = sub + 16 * j;
    dst[idx] = (bf16)src[(size_t)k * MD + o];
}

// out[n] = bias[n] + sum_k v[k] * W[k*MD + n]
__global__ void k_vecmat(const float* __restrict__ v, const float* __restrict__ W,
                         const float* __restrict__ bias, float* __restrict__ out, int K) {
    int n = blockIdx.x * blockDim.x + threadIdx.x;
    float acc = bias ? bias[n] : 0.0f;
    for (int k = 0; k < K; ++k) acc += v[k] * W[(size_t)k * MD + n];
    out[n] = acc;
}

// X[t,n] = gelu( sum_s ratios[t,s]*P[s,n] + b[n] ) ; also bf16 copy
__global__ void k_sp_pre(const float* __restrict__ ratios, const float* __restrict__ P,
                         const float* __restrict__ b, float* __restrict__ X,
                         bf16* __restrict__ Xb) {
    int idx = blockIdx.x * blockDim.x + threadIdx.x;
    int t = idx / MD, n = idx % MD;
    float acc = b[n];
#pragma unroll
    for (int s = 0; s < NSOL; ++s) acc += ratios[t * NSOL + s] * P[s * MD + n];
    float g = geluf(acc);
    X[idx] = g;
    Xb[idx] = (bf16)g;
}

__global__ void k_demb(const float* __restrict__ desc, const float* __restrict__ w,
                       const float* __restrict__ b, float* __restrict__ demb) {
    int idx = blockIdx.x * blockDim.x + threadIdx.x;
    int t = idx / DE, j = idx % DE;
    float acc = b[j];
#pragma unroll
    for (int c = 0; c < 6; ++c) acc += desc[t * 6 + c] * w[c * DE + j];
    demb[idx] = geluf(acc);
}

__global__ void k_base0(const float* __restrict__ demb, const float* __restrict__ Wc,
                        const float* __restrict__ molA, float* __restrict__ base) {
    int idx = blockIdx.x * blockDim.x + threadIdx.x;
    int t = idx / MD, n = idx % MD;
    float acc = molA[n];
    for (int j = 0; j < DE; ++j) acc += demb[t * DE + j] * Wc[(size_t)j * MD + n];
    base[idx] = acc;
}

// ---------------------------------------------------------------------------
// WMMA bf16 GEMM (batched-over-T phase), modes as before
// ---------------------------------------------------------------------------
__device__ __forceinline__ v16bf load_frag(const bf16* __restrict__ p, int r0,
                                           int kb, int lane) {
    int half = lane >> 4, m = lane & 15;
    const bf16* q = p + (size_t)(r0 + m) * MD + kb + 8 * half;
    v8bf lo = *(const v8bf*)q;
    v8bf hi = *(const v8bf*)(q + 16);
    v16bf f;
#pragma unroll
    for (int i = 0; i < 8; ++i) { f[i] = lo[i]; f[8 + i] = hi[i]; }
    return f;
}

__global__ void __launch_bounds__(256)
k_gemm(const bf16* __restrict__ A, const bf16* __restrict__ Bt,
       const float* __restrict__ bias, const float* __restrict__ R,
       bf16* __restrict__ OB, float* __restrict__ OF, int mode) {
    int lane = threadIdx.x & 31, wave = threadIdx.x >> 5;
    int row0 = blockIdx.x * 32 + (wave >> 2) * 16;
    int col0 = blockIdx.y * 64 + (wave & 3) * 16;
    v8f c = {};
    for (int kb = 0; kb < MD; kb += 32) {
        v16bf a = load_frag(A, row0, kb, lane);
        v16bf b = load_frag(Bt, col0, kb, lane);
        c = __builtin_amdgcn_wmma_f32_16x16x32_bf16(false, a, false, b,
                                                    (short)0, c, false, false);
    }
    int n = col0 + (lane & 15);
    int mbase = row0 + 8 * (lane >> 4);
#pragma unroll
    for (int r = 0; r < 8; ++r) {
        size_t o = (size_t)(mbase + r) * MD + n;
        float v = c[r];
        if (mode == 0)      OB[o] = (bf16)geluf(v + bias[n]);
        else if (mode == 1) OF[o] = v + bias[n] + R[o];
        else                OF[o] += v;
    }
}

// Row LayerNorm (batched phase)
__global__ void __launch_bounds__(256)
k_ln(const float* __restrict__ Y, const float* __restrict__ g,
     const float* __restrict__ b, float* __restrict__ X, bf16* __restrict__ Xb) {
    __shared__ float red[2][8];
    int t = blockIdx.x;
    const float* y = Y + (size_t)t * MD;
    float s = 0.0f, s2 = 0.0f;
    for (int n = threadIdx.x; n < MD; n += 256) { float v = y[n]; s += v; s2 += v * v; }
    for (int off = 16; off; off >>= 1) {
        s  += __shfl_down(s,  off, 32);
        s2 += __shfl_down(s2, off, 32);
    }
    int wave = threadIdx.x >> 5, lane = threadIdx.x & 31;
    if (!lane) { red[0][wave] = s; red[1][wave] = s2; }
    __syncthreads();
    s = 0.0f; s2 = 0.0f;
#pragma unroll
    for (int i = 0; i < 8; ++i) { s += red[0][i]; s2 += red[1][i]; }
    float mean = s * (1.0f / MD);
    float inv  = rsqrtf(s2 * (1.0f / MD) - mean * mean + 1e-5f);
    for (int n = threadIdx.x; n < MD; n += 256) {
        float v = (y[n] - mean) * inv * g[n] + b[n];
        size_t o = (size_t)t * MD + n;
        X[o] = v;
        Xb[o] = (bf16)v;
    }
}

// ---------------------------------------------------------------------------
// Device-scope generation barrier across the NWG persistent workgroups
// ---------------------------------------------------------------------------
__device__ __forceinline__ void global_barrier(int* cnt, int* gen) {
    __syncthreads();
    if (threadIdx.x == 0) {
        __threadfence();
        int g = __hip_atomic_load(gen, __ATOMIC_RELAXED, __HIP_MEMORY_SCOPE_AGENT);
        int a = __hip_atomic_fetch_add(cnt, 1, __ATOMIC_ACQ_REL, __HIP_MEMORY_SCOPE_AGENT);
        if (a == NWG - 1) {
            __hip_atomic_store(cnt, 0, __ATOMIC_RELAXED, __HIP_MEMORY_SCOPE_AGENT);
            __hip_atomic_fetch_add(gen, 1, __ATOMIC_ACQ_REL, __HIP_MEMORY_SCOPE_AGENT);
        } else {
            while (__hip_atomic_load(gen, __ATOMIC_ACQUIRE, __HIP_MEMORY_SCOPE_AGENT) == g)
                __builtin_amdgcn_s_sleep(1);
        }
    }
    __syncthreads();
}

__global__ void k_init(int* cnt, int* gen) {
    if (threadIdx.x == 0) { *cnt = 0; *gen = 0; }
}

// ---------------------------------------------------------------------------
// Multi-workgroup sequential scan: NWG x 256 threads, each WG owns OPW=16 rows
// of every beta-head weight matrix. 6 global barriers per timestep; LayerNorm
// partials go to write-once slots; rf epilogue computed redundantly per WG.
// ---------------------------------------------------------------------------
__global__ void __launch_bounds__(256)
k_scan_mwg(const float* __restrict__ base, const float* __restrict__ wlast,
           const unsigned char* __restrict__ bmask,
           const bf16* __restrict__ W1t, const float* __restrict__ b1,
           const bf16* __restrict__ W2t, const float* __restrict__ b2,
           const float* __restrict__ lng, const float* __restrict__ lnb,
           const float* __restrict__ muw, const float* __restrict__ mub,
           const float* __restrict__ phiw, const float* __restrict__ phib,
           float* __restrict__ hbuf, float* __restrict__ ybuf,
           float* __restrict__ parts, int* cnt, int* gen,
           float* __restrict__ out) {
    __shared__ float zin[MD];        // full input activation of current block
    __shared__ float hl[MD];         // full h vector (copied after h-barrier)
    __shared__ float zloc[OPW];      // this WG's residual terms
    __shared__ float yred[2][OPW];   // per-WG LN partial staging
    __shared__ float pr[2][NWG];     // gathered LN partials
    __shared__ float sred[2][8];
    __shared__ float s_stats[2];     // mean, inv
    __shared__ float s_prev;

    const int tid = threadIdx.x;
    const int wg  = blockIdx.x;
    const int o   = tid >> 4;        // local output row 0..15
    const int sub = tid & 15;        // k-chunk index
    const int oo  = wg * OPW + o;    // global output row

    if (tid == 0) s_prev = -1.0f;
    __syncthreads();

    for (int t = 0; t < TT; ++t) {
        const float prev = s_prev;
        for (int blk = 0; blk < NB; ++blk) {
            // ---- build zin (block input) in LDS ----
            if (blk == 0) {
                for (int k = tid; k < MD; k += 256)
                    zin[k] = geluf(base[(size_t)t * MD + k] + prev * wlast[k]);
            } else {
                if (tid < NWG) {
                    size_t s0 = ((size_t)(t * NB + (blk - 1)) * NWG + tid) * 2;
                    pr[0][tid] = gload(parts + s0);
                    pr[1][tid] = gload(parts + s0 + 1);
                }
                __syncthreads();
                if (tid == 0) {
                    float s = 0.0f, s2 = 0.0f;
                    for (int w = 0; w < NWG; ++w) { s += pr[0][w]; s2 += pr[1][w]; }
                    float mean = s * (1.0f / MD);
                    s_stats[0] = mean;
                    s_stats[1] = rsqrtf(s2 * (1.0f / MD) - mean * mean + 1e-5f);
                }
                __syncthreads();
                float mean = s_stats[0], inv = s_stats[1];
                const float* g  = lng + (blk - 1) * MD;
                const float* bb = lnb + (blk - 1) * MD;
                for (int k = tid; k < MD; k += 256)
                    zin[k] = (gload(ybuf + k) - mean) * inv * g[k] + bb[k];
            }
            __syncthreads();
            if (sub == 0) zloc[o] = zin[oo];

            // ---- h = gelu(zin @ W1 + b1), sharded: 16 threads per output ----
            {
                const v8bf* w8 = (const v8bf*)(W1t + ((size_t)blk * MD + oo) * MD + sub * 64);
                float acc = 0.0f;
#pragma unroll
                for (int q = 0; q < 8; ++q) {
                    v8bf w = w8[q];
#pragma unroll
                    for (int u = 0; u < 8; ++u)
                        acc += (float)w[u] * zin[sub + 16 * (q * 8 + u)];
                }
                for (int d = 8; d; d >>= 1) acc += __shfl_down(acc, d, 16);
                if (sub == 0) gstore(hbuf + oo, geluf(acc + b1[blk * MD + oo]));
            }
            global_barrier(cnt, gen);                       // h ready everywhere

            for (int k = tid; k < MD; k += 256) hl[k] = gload(hbuf + k);
            __syncthreads();

            // ---- y = h @ W2 + b2 + z ; publish y and LN partials ----
            {
                const v8bf* w8 = (const v8bf*)(W2t + ((size_t)blk * MD + oo) * MD + sub * 64);
                float acc = 0.0f;
#pragma unroll
                for (int q = 0; q < 8; ++q) {
                    v8bf w = w8[q];
#pragma unroll
                    for (int u = 0; u < 8; ++u)
                        acc += (float)w[u] * hl[sub + 16 * (q * 8 + u)];
                }
                for (int d = 8; d; d >>= 1) acc += __shfl_down(acc, d, 16);
                if (sub == 0) {
                    float y = acc + b2[blk * MD + oo] + zloc[o];
                    gstore(ybuf + oo, y);
                    yred[0][o] = y;
                    yred[1][o] = y * y;
                }
            }
            __syncthreads();
            if (tid == 0) {
                float s = 0.0f, s2 = 0.0f;
#pragma unroll
                for (int i = 0; i < OPW; ++i) { s += yred[0][i]; s2 += yred[1][i]; }
                size_t s0 = ((size_t)(t * NB + blk) * NWG + wg) * 2;
                gstore(parts + s0, s);
                gstore(parts + s0 + 1, s2);
            }
            global_barrier(cnt, gen);                       // y + partials ready
        }

        // ---- epilogue (redundant in every WG): z3 = LN(y3), mu/phi, rf ----
        if (tid < NWG) {
            size_t s0 = ((size_t)(t * NB + (NB - 1)) * NWG + tid) * 2;
            pr[0][tid] = gload(parts + s0);
            pr[1][tid] = gload(parts + s0 + 1);
        }
        __syncthreads();
        if (tid == 0) {
            float s = 0.0f, s2 = 0.0f;
            for (int w = 0; w < NWG; ++w) { s += pr[0][w]; s2 += pr[1][w]; }
            float mean = s * (1.0f / MD);
            s_stats[0] = mean;
            s_stats[1] = rsqrtf(s2 * (1.0f / MD) - mean * mean + 1e-5f);
        }
        __syncthreads();
        {
            float mean = s_stats[0], inv = s_stats[1];
            const float* g  = lng + (NB - 1) * MD;
            const float* bb = lnb + (NB - 1) * MD;
            float dm = 0.0f, dp = 0.0f;
            for (int k = tid; k < MD; k += 256) {
                float z = (gload(ybuf + k) - mean) * inv * g[k] + bb[k];
                dm += z * muw[k];
                dp += z * phiw[k];
            }
            for (int d = 16; d; d >>= 1) {
                dm += __shfl_down(dm, d, 32);
                dp += __shfl_down(dp, d, 32);
            }
            int wv = tid >> 5, ln = tid & 31;
            if (!ln) { sred[0][wv] = dm; sred[1][wv] = dp; }
            __syncthreads();
            if (tid == 0) {
                float sm = 0.0f, sp = 0.0f;
#pragma unroll
                for (int w = 0; w < 8; ++w) { sm += sred[0][w]; sp += sred[1][w]; }
                float mu = 1.0f / (1.0f + expf(-(sm + mub[0])));
                float ph = sp + phib[0];
                ph = (ph > 20.0f ? ph : log1pf(expf(ph))) + 2.0f;
                float rf = (bmask[t] != 0 || prev < 0.0f) ? mu : prev + mu * (1.0f - prev);
                if (wg == 0) { out[t] = rf; out[TT + t] = mu; out[2 * TT + t] = ph; }
                s_prev = rf;
            }
            __syncthreads();
        }
    }
}

// ---------------------------------------------------------------------------
extern "C" void kernel_launch(void* const* d_in, const int* in_sizes, int n_in,
                              void* d_out, int out_size, void* d_ws, size_t ws_size,
                              hipStream_t stream) {
    (void)in_sizes; (void)n_in; (void)out_size; (void)ws_size;
    const float* mol_vec   = (const float*)d_in[0];
    const float* ratios    = (const float*)d_in[1];
    const float* desc_seq  = (const float*)d_in[2];
    const float* solvent   = (const float*)d_in[3];
    const unsigned char* bmask = (const unsigned char*)d_in[4];
    const float* sp_proj_w = (const float*)d_in[5];
    const float* sp_proj_b = (const float*)d_in[6];
    const float* sp_fc1_w  = (const float*)d_in[7];
    const float* sp_fc1_b  = (const float*)d_in[8];
    const float* sp_fc2_w  = (const float*)d_in[9];
    const float* sp_fc2_b  = (const float*)d_in[10];
    const float* sp_ln_g   = (const float*)d_in[11];
    const float* sp_ln_b   = (const float*)d_in[12];
    const float* desc_w    = (const float*)d_in[13];
    const float* desc_b    = (const float*)d_in[14];
    const float* bh_proj_w = (const float*)d_in[15];
    const float* bh_proj_b = (const float*)d_in[16];
    const float* bh_fc1_w  = (const float*)d_in[17];
    const float* bh_fc1_b  = (const float*)d_in[18];
    const float* bh_fc2_w  = (const float*)d_in[19];
    const float* bh_fc2_b  = (const float*)d_in[20];
    const float* bh_ln_g   = (const float*)d_in[21];
    const float* bh_ln_b   = (const float*)d_in[22];
    const float* mu_w      = (const float*)d_in[23];
    const float* mu_b      = (const float*)d_in[24];
    const float* phi_w     = (const float*)d_in[25];
    const float* phi_b     = (const float*)d_in[26];

    char* ws = (char*)d_ws;
    size_t off = 0;
    auto carve = [&](size_t bytes) -> char* {
        char* p = ws + off;
        off = (off + bytes + 255) & ~(size_t)255;
        return p;
    };
    bf16*  sp1t  = (bf16*)carve(NB * MM * sizeof(bf16));
    bf16*  sp2t  = (bf16*)carve(NB * MM * sizeof(bf16));
    bf16*  bh1t  = (bf16*)carve(NB * MM * sizeof(bf16));     // scan-permuted
    bf16*  bh2t  = (bf16*)carve(NB * MM * sizeof(bf16));     // scan-permuted
    bf16*  Wbt   = (bf16*)carve(MM * sizeof(bf16));
    float* P     = (float*)carve(NSOL * MD * sizeof(float));
    float* molA  = (float*)carve(MD * sizeof(float));
    float* X     = (float*)carve((size_t)TT * MD * sizeof(float));
    bf16*  Xb    = (bf16*)carve((size_t)TT * MD * sizeof(bf16));
    bf16*  Hb    = (bf16*)carve((size_t)TT * MD * sizeof(bf16));
    float* Y     = (float*)carve((size_t)TT * MD * sizeof(float));
    float* demb  = (float*)carve((size_t)TT * DE * sizeof(float));
    float* base  = (float*)carve((size_t)TT * MD * sizeof(float));
    float* hbuf  = (float*)carve(MD * sizeof(float));
    float* ybuf  = (float*)carve(MD * sizeof(float));
    float* parts = (float*)carve((size_t)TT * NB * NWG * 2 * sizeof(float));
    int*   bar   = (int*)carve(2 * sizeof(int));

    dim3 tb(32, 8), tg(32, 32);
    for (int b = 0; b < NB; ++b) {
        k_cvt_t<<<tg, tb, 0, stream>>>(sp_fc1_w + (size_t)b * MM, sp1t + (size_t)b * MM);
        k_cvt_t<<<tg, tb, 0, stream>>>(sp_fc2_w + (size_t)b * MM, sp2t + (size_t)b * MM);
        k_cvt_perm<<<MM / 256, 256, 0, stream>>>(bh_fc1_w + (size_t)b * MM, bh1t + (size_t)b * MM);
        k_cvt_perm<<<MM / 256, 256, 0, stream>>>(bh_fc2_w + (size_t)b * MM, bh2t + (size_t)b * MM);
    }
    k_cvt_t<<<tg, tb, 0, stream>>>(bh_proj_w + MM, Wbt);     // rows 1024..2047

    k_vecmat<<<MD / 256, 256, 0, stream>>>(mol_vec, bh_proj_w, bh_proj_b, molA, MD);
    for (int s = 0; s < NSOL; ++s)
        k_vecmat<<<MD / 256, 256, 0, stream>>>(solvent + (size_t)s * MD,
                                               sp_proj_w + (size_t)s * MM,
                                               nullptr, P + s * MD, MD);

    k_sp_pre<<<(TT * MD) / 256, 256, 0, stream>>>(ratios, P, sp_proj_b, X, Xb);
    dim3 gg(TT / 32, MD / 64);
    for (int b = 0; b < NB; ++b) {
        k_gemm<<<gg, 256, 0, stream>>>(Xb, sp1t + (size_t)b * MM, sp_fc1_b + b * MD,
                                       nullptr, Hb, nullptr, 0);
        k_gemm<<<gg, 256, 0, stream>>>(Hb, sp2t + (size_t)b * MM, sp_fc2_b + b * MD,
                                       X, nullptr, Y, 1);
        k_ln<<<TT, 256, 0, stream>>>(Y, sp_ln_g + b * MD, sp_ln_b + b * MD, X, Xb);
    }

    k_demb<<<(TT * DE) / 256, 256, 0, stream>>>(desc_seq, desc_w, desc_b, demb);
    k_base0<<<(TT * MD) / 256, 256, 0, stream>>>(demb, bh_proj_w + (size_t)(2 * MD) * MD,
                                                 molA, base);
    k_gemm<<<gg, 256, 0, stream>>>(Xb, Wbt, nullptr, nullptr, nullptr, base, 2);

    k_init<<<1, 32, 0, stream>>>(bar, bar + 1);
    k_scan_mwg<<<NWG, 256, 0, stream>>>(base, bh_proj_w + (size_t)(2 * MD + DE) * MD, bmask,
                                        bh1t, bh_fc1_b, bh2t, bh_fc2_b, bh_ln_g, bh_ln_b,
                                        mu_w, mu_b, phi_w, phi_b,
                                        hbuf, ybuf, parts, bar, bar + 1,
                                        (float*)d_out);
}
